// AgentNetwork_46720654246425
// MI455X (gfx1250) — matrix operand order repaired and checked by reading, not compiled, not run on hardware
//
#include <hip/hip_runtime.h>

typedef __attribute__((ext_vector_type(16))) _Float16 v16h;
typedef __attribute__((ext_vector_type(8)))  _Float16 v8h;
typedef __attribute__((ext_vector_type(2)))  __fp16   v2hp;  // matches cvt_pkrtz return type
typedef __attribute__((ext_vector_type(8)))  float    v8f;

// Problem constants
constexpr int B = 512, S = 512, I = 64, C = 4, H = 100, O = 7;
constexpr int IC = I + C;        // 68
constexpr int KX = 96;           // padded input K  (3 * 32)
constexpr int KH = 128;          // padded hidden K (4 * 32)
constexpr int NP = 128;          // padded N        (8 * 16) -> 4 static tiles per wave
constexpr int QT = 4;            // N tiles per wave (wave0: 0..3, wave1: 4..7; tile 7 = dummy)
constexpr int BT = 16;           // batch tile per block

// Dynamic LDS layout (byte offsets, all 64B aligned)
constexpr int OFF_WFX = 0;                      // [100][KX] f16 (zeros k>=68)
constexpr int OFF_WHX = OFF_WFX + 100 * KX * 2; // 19200
constexpr int OFF_WFH = OFF_WHX + 100 * KX * 2; // 38400  [100][KH] f16 (zeros k>=100)
constexpr int OFF_WHH = OFF_WFH + 100 * KH * 2; // 64000
constexpr int OFF_H   = OFF_WHH + 100 * KH * 2; // 89600  [BT][KH] f16
constexpr int OFF_FH  = OFF_H   + BT * KH * 2;  // 93696  [BT][KH] f16
constexpr int OFF_XC  = OFF_FH  + BT * KH * 2;  // 97792  [BT][KX] f16 (ctx cols 64..67, zeros 68..95)
constexpr int OFF_RDO = OFF_XC  + BT * KX * 2;  // 100864 [BT][NP] f32 (final readout staging)
constexpr int OFF_WFB = OFF_RDO + BT * NP * 4;  // 109056 [NP] f32 (zeros >= 100)
constexpr int OFF_WHB = OFF_WFB + NP * 4;       // 109568
constexpr int OFF_ROW = OFF_WHB + NP * 4;       // 110080 [O][H] f32 (pad to 2816)
constexpr int OFF_ROB = OFF_ROW + 2816;         // 112896
constexpr int SMEM_BYTES = OFF_ROB + 64;        // 112960 (~110 KB of 320 KB WGP LDS)

__device__ __forceinline__ v8f zero8() {
    v8f z = {0.f, 0.f, 0.f, 0.f, 0.f, 0.f, 0.f, 0.f};
    return z;
}

__device__ __forceinline__ v8f wmma_f16(v16h a, v16h b, v8f c) {
    // D = A(16x32 f16) * B(32x16 f16) + C(16x16 f32)
    return __builtin_amdgcn_wmma_f32_16x16x32_f16(
        /*neg_a=*/false, a, /*neg_b=*/false, b,
        /*c_mod=*/(short)0, c, /*reuse_a=*/false, /*reuse_b=*/false);
}

// A-fragment gather: lane<16 holds K {0..7,16..23}, lane>=16 holds K {8..15,24..31}
__device__ __forceinline__ v16h load_Afrag(const _Float16* p) {
    v8h lo = *(const v8h*)(p);
    v8h hi = *(const v8h*)(p + 16);
    return __builtin_shufflevector(lo, hi, 0,1,2,3,4,5,6,7,8,9,10,11,12,13,14,15);
}

// B-fragment: lane<16 holds K {0..7,16..23}, lane>=16 holds K {8..15,24..31} of one column
__device__ __forceinline__ v16h load_Bfrag(const _Float16* p) {
    v8h lo = *(const v8h*)(p);
    v8h hi = *(const v8h*)(p + 16);
    return __builtin_shufflevector(lo, hi, 0,1,2,3,4,5,6,7,8,9,10,11,12,13,14,15);
}

__device__ __forceinline__ float fast_sigmoid(float z) {
    return __builtin_amdgcn_rcpf(1.0f + __expf(-z));   // raw v_rcp_f32
}
__device__ __forceinline__ float fast_tanh(float z) {
    const float e = __expf(-2.0f * z);
    return (1.0f - e) * __builtin_amdgcn_rcpf(1.0f + e);
}

__global__ __launch_bounds__(64) void agent_rnn_kernel(
    const float* __restrict__ x,     const float* __restrict__ ctx,
    const float* __restrict__ Wf_w,  const float* __restrict__ Wf_b,
    const float* __restrict__ Wh_w,  const float* __restrict__ Wh_b,
    const float* __restrict__ ro_w,  const float* __restrict__ ro_b,
    float* __restrict__ out,         float* __restrict__ hs)
{
    extern __shared__ char smem[];
    _Float16* WFX  = (_Float16*)(smem + OFF_WFX);
    _Float16* WHX  = (_Float16*)(smem + OFF_WHX);
    _Float16* WFH  = (_Float16*)(smem + OFF_WFH);
    _Float16* WHH  = (_Float16*)(smem + OFF_WHH);
    _Float16* HLDS = (_Float16*)(smem + OFF_H);
    _Float16* FHLD = (_Float16*)(smem + OFF_FH);
    _Float16* XCL  = (_Float16*)(smem + OFF_XC);
    float*    RDO  = (float*)(smem + OFF_RDO);
    float*    WFB  = (float*)(smem + OFF_WFB);
    float*    WHB  = (float*)(smem + OFF_WHB);
    float*    ROW  = (float*)(smem + OFF_ROW);
    float*    ROB  = (float*)(smem + OFF_ROB);

    const int tid  = threadIdx.x;
    const int lane = tid & 31;
    const int wv   = tid >> 5;
    const int b0   = blockIdx.x * BT;

    // ---------------- setup: weights / biases / staging pads into LDS ----------------
    for (int i = tid; i < 100 * KX; i += 64) {
        int n = i / KX, k = i - n * KX;
        float fv = (k < IC) ? Wf_w[n * (IC + H) + k] : 0.f;
        float hv = (k < IC) ? Wh_w[n * (IC + H) + k] : 0.f;
        WFX[i] = (_Float16)fv;
        WHX[i] = (_Float16)hv;
    }
    for (int i = tid; i < 100 * KH; i += 64) {
        int n = i / KH, k = i - n * KH;
        float fv = (k < H) ? Wf_w[n * (IC + H) + IC + k] : 0.f;
        float hv = (k < H) ? Wh_w[n * (IC + H) + IC + k] : 0.f;
        WFH[i] = (_Float16)fv;   // exact zeros for k in [100,128): pad columns annihilate
        WHH[i] = (_Float16)hv;
    }
    for (int i = tid; i < NP; i += 64) {
        WFB[i] = (i < H) ? Wf_b[i] : 0.f;
        WHB[i] = (i < H) ? Wh_b[i] : 0.f;
    }
    for (int i = tid; i < O * H; i += 64) ROW[i] = ro_w[i];
    if (tid < O) ROB[tid] = ro_b[tid];
    for (int i = tid; i < BT * KH; i += 64) {   // h0 = 0
        HLDS[i] = (_Float16)0.f;
        FHLD[i] = (_Float16)0.f;
    }
    for (int i = tid; i < BT * KX; i += 64) {   // context cols + zero padding
        int c = i % KX;
        float v = (c >= I && c < IC) ? ctx[c - I] : 0.f;
        XCL[i] = (_Float16)v;
    }
    __syncthreads();

    // ---------------- t-invariant per-lane state (hoisted out of the scan) ----------------
    const int hi  = lane >> 4;     // half-wave select
    const int nlo = lane & 15;
    const int bk  = hi * 8;        // B-fragment K sub-offset
    const int ak  = hi * 8;        // A-fragment K sub-offset

    v8f hreg[QT];                  // hidden state, register-resident
    #pragma unroll
    for (int q = 0; q < QT; ++q) hreg[q] = zero8();

    int   nq[QT];
    const _Float16 *pWFX[QT], *pWHX[QT], *pWFH[QT], *pWHH[QT];
    _Float16 *pFH[QT], *pHL[QT];   // per-q lane store pointers (r folds into ds immediate)
    float biasF[QT], biasH[QT];
    #pragma unroll
    for (int q = 0; q < QT; ++q) {
        const int nt = wv * QT + q;
        const int n  = nt * 16 + nlo;
        const int nc = (n < H) ? n : (H - 1);   // clamp: garbage only lands in discarded cols
        nq[q]    = n;
        pWFX[q]  = WFX + nc * KX + bk;
        pWHX[q]  = WHX + nc * KX + bk;
        pWFH[q]  = WFH + nc * KH + bk;
        pWHH[q]  = WHH + nc * KH + bk;
        pFH[q]   = FHLD + hi * 8 * KH + n;
        pHL[q]   = HLDS + hi * 8 * KH + n;
        biasF[q] = WFB[n];
        biasH[q] = WHB[n];
    }
    const _Float16* aXC = XCL  + nlo * KX + ak;
    const _Float16* aH  = HLDS + nlo * KH + ak;
    const _Float16* aFH = FHLD + nlo * KH + ak;

    const int xrow = tid >> 2;     // x staging: 4 lanes per batch row, 16 f32 each
    const int xq   = tid & 3;
    const float* xbase = x + ((size_t)(b0 + xrow) * S) * (size_t)I + xq * 16;
    _Float16*    xcdst = XCL + xrow * KX + xq * 16;

    // running per-lane output-row pointer: hs[(b0 + hi*8 + r)][t][...] ; r*S*H folds to imm
    float* hsrow = hs + (size_t)(b0 + hi * 8) * S * (size_t)H;

    for (int t = 0; t < S; ++t) {
        // ---- stage x tile (16 x 64 f32 -> packed f16 in LDS), prefetch next step ----
        {
            const float* xp = xbase + (size_t)t * I;
            #pragma unroll
            for (int j = 0; j < 4; ++j) {
                float4 v = *(const float4*)(xp + j * 4);
                *(v2hp*)(xcdst + j * 4)     = __builtin_amdgcn_cvt_pkrtz(v.x, v.y);
                *(v2hp*)(xcdst + j * 4 + 2) = __builtin_amdgcn_cvt_pkrtz(v.z, v.w);
            }
            if (t + 1 < S) __builtin_prefetch(xp + I, 0, 1);  // global_prefetch_b8
        }
        __syncthreads();  // xc ready; previous step's h writes visible

        // ---- gather A fragments for xc (16x96) and h (16x128) ----
        v16h Axc[3], Ah[4];
        #pragma unroll
        for (int kt = 0; kt < 3; ++kt) Axc[kt] = load_Afrag(aXC + kt * 32);
        #pragma unroll
        for (int kt = 0; kt < 4; ++kt) Ah[kt] = load_Afrag(aH + kt * 32);

        // ---- pass 1: f = sigmoid(xc@WfxT + h@WfhT + bf); keep f and xh-part in regs ----
        v8f fsv[QT], xhv[QT];
        #pragma unroll
        for (int q = 0; q < QT; ++q) {
            // three independent WMMA chains (depth 3 / 4 / 3) instead of one depth-7 chain
            v8f accFx = zero8(), accFh = zero8(), accH = zero8();
            #pragma unroll
            for (int kt = 0; kt < 3; ++kt) {
                accFx = wmma_f16(Axc[kt], load_Bfrag(pWFX[q] + kt * 32), accFx);
                accH  = wmma_f16(Axc[kt], load_Bfrag(pWHX[q] + kt * 32), accH);
            }
            #pragma unroll
            for (int kt = 0; kt < 4; ++kt) {
                accFh = wmma_f16(Ah[kt], load_Bfrag(pWFH[q] + kt * 32), accFh);
            }
            xhv[q] = accH;
            #pragma unroll
            for (int r = 0; r < 8; ++r) {
                const float f = fast_sigmoid(accFx[r] + accFh[r] + biasF[q]);
                fsv[q][r] = f;
                // unguarded: cols >=100 hold finite garbage, annihilated by zero B rows
                pFH[q][r * KH] = (_Float16)(f * hreg[q][r]);
            }
        }
        __syncthreads();  // f*h staged for all columns

        // ---- pass 2: h_tilde = tanh(xh + (f*h)@WhhT + bh); h = (1-f)h + f*h_tilde ----
        v16h Afh[4];
        #pragma unroll
        for (int kt = 0; kt < 4; ++kt) Afh[kt] = load_Afrag(aFH + kt * 32);

        #pragma unroll
        for (int q = 0; q < QT; ++q) {
            v8f acc = zero8();
            #pragma unroll
            for (int kt = 0; kt < 4; ++kt) {
                acc = wmma_f16(Afh[kt], load_Bfrag(pWHH[q] + kt * 32), acc);
            }
            float hn8[8];
            #pragma unroll
            for (int r = 0; r < 8; ++r) {
                const float ht = fast_tanh(xhv[q][r] + acc[r] + biasH[q]);
                const float f  = fsv[q][r];
                const float hn = (1.0f - f) * hreg[q][r] + f * ht;
                hreg[q][r] = hn;
                hn8[r] = hn;
                pHL[q][r * KH] = (_Float16)hn;   // unguarded (see pass-1 note)
            }
            if (nq[q] < H) {                     // single exec toggle per tile
                float* p = hsrow + nq[q];
                #pragma unroll
                for (int r = 0; r < 8; ++r)
                    __builtin_nontemporal_store(hn8[r], p + (size_t)r * S * H);
            }
        }
        hsrow += H;  // advance to next timestep
        // next iteration's __syncthreads separates HLDS writes from A_h gathers
    }

    // ---------------- readout: out = h_last @ ro_w.T + ro_b ----------------
    #pragma unroll
    for (int q = 0; q < QT; ++q) {
        const int n = nq[q];
        #pragma unroll
        for (int r = 0; r < 8; ++r) RDO[(r + hi * 8) * NP + n] = hreg[q][r];
    }
    __syncthreads();
    if (tid < BT) {
        const int m = tid;
        #pragma unroll
        for (int o = 0; o < O; ++o) {
            float acc = ROB[o];
            for (int k = 0; k < H; ++k) acc += RDO[m * NP + k] * ROW[o * H + k];
            out[(size_t)(b0 + m) * O + o] = acc;
        }
    }
}

extern "C" void kernel_launch(void* const* d_in, const int* in_sizes, int n_in,
                              void* d_out, int out_size, void* d_ws, size_t ws_size,
                              hipStream_t stream) {
    (void)in_sizes; (void)n_in; (void)d_ws; (void)ws_size; (void)out_size;
    const float* x    = (const float*)d_in[0];
    const float* ctx  = (const float*)d_in[1];
    const float* Wf_w = (const float*)d_in[2];
    const float* Wf_b = (const float*)d_in[3];
    const float* Wh_w = (const float*)d_in[4];
    const float* Wh_b = (const float*)d_in[5];
    const float* ro_w = (const float*)d_in[6];
    const float* ro_b = (const float*)d_in[7];

    float* out = (float*)d_out;            // (B, O) first
    float* hsp = out + (size_t)B * O;      // then (B, S, H)

    // ~110 KB dynamic LDS per workgroup (WGP has 320 KB on gfx1250)
    (void)hipFuncSetAttribute((const void*)agent_rnn_kernel,
                              hipFuncAttributeMaxDynamicSharedMemorySize, SMEM_BYTES);

    agent_rnn_kernel<<<dim3(B / BT), dim3(64), SMEM_BYTES, stream>>>(
        x, ctx, Wf_w, Wf_b, Wh_w, Wh_b, ro_w, ro_b, out, hsp);
}
